// QuatNet_8675833938067
// MI455X (gfx1250) — matrix operand order repaired and checked by reading, not compiled, not run on hardware
//
#include <hip/hip_runtime.h>

typedef unsigned int u32;
typedef __attribute__((ext_vector_type(16))) __bf16 v16bf;
typedef __attribute__((ext_vector_type(8)))  float  v8f;
typedef __attribute__((ext_vector_type(8)))  u32    u32x8;
typedef __attribute__((ext_vector_type(4)))  float  f32x4;
typedef __attribute__((ext_vector_type(2)))  float  f32x2;

#define NB   131072
#define ND   512
#define NH   2048
#define NOUT 10

// workspace layout (dwords)
#define W1_DW   524288          // 16 kc * 128 ntiles * 32 lanes * 8 dwords
#define W2_DW   16384           // 64 kc * 1 ntile  * 32 lanes * 8 dwords
#define OFF_W1H 0
#define OFF_W1L (W1_DW)
#define OFF_W2H (2*W1_DW)
#define OFF_W2L (2*W1_DW + W2_DW)

// LDS layout (dwords), 4 waves per WG
#define LDS_A_PER_WAVE   16384  // 2 mfrag * 16 kc * 2 part * 32 lane * 8 dw = 64KB
#define LDS_STG_BASE     65536  // + w*1024 : 32x32 f32 stage per wave
#define LDS_C2_BASE      69632  // + w*512  : 32x16 f32 per wave
#define LDS_TOTAL_DW     71680  // 280 KB of 320 KB WGP LDS
#define LDS_BYTES        (LDS_TOTAL_DW*4)

static __device__ __forceinline__ u32 pack_hi(float a, float b) {
  unsigned short ha = __builtin_bit_cast(unsigned short, (__bf16)a);
  unsigned short hb = __builtin_bit_cast(unsigned short, (__bf16)b);
  return (u32)ha | ((u32)hb << 16);
}
static __device__ __forceinline__ u32 pack_lo(float a, float b) {
  float ra = a - (float)(__bf16)a;
  float rb = b - (float)(__bf16)b;
  return pack_hi(ra, rb);
}
static __device__ __forceinline__ v8f wmma_bf16(u32x8 a, u32x8 b, v8f c) {
  return __builtin_amdgcn_wmma_f32_16x16x32_bf16(
      false, __builtin_bit_cast(v16bf, a),
      false, __builtin_bit_cast(v16bf, b),
      (short)0, c, false, false);
}
static __device__ __forceinline__ void wave_lds_fence() {
  // same-wave LDS ops are in-order; this is a belt-and-braces wait + compiler
  // memory barrier between the LDS stage store and its re-read.
  asm volatile("s_wait_dscnt 0" ::: "memory");
}

// ---------------- prep: pack W1 into bf16 hi/lo B-fragment tiles ------------
// B (32x16, 16-bit) layout per ISA 05_wmma: VGPR v of lane holds
// K = (v%4)*2 + (lane/16)*8 + (v/4)*16 (+0,+1), n = lane%16.
__global__ __launch_bounds__(256) void prep_w1(const float* __restrict__ W1,
                                               u32* __restrict__ hi,
                                               u32* __restrict__ lo) {
  int t = blockIdx.x * 256 + threadIdx.x;        // 524288 threads
  int frag = t >> 8, pos = t & 255;
  int lane = pos >> 3, v = pos & 7;
  int kc = frag >> 7, nt = frag & 127;
  int n = nt * 16 + (lane & 15);
  int k = kc * 32 + (v & 3) * 2 + (lane >> 4) * 8 + (v >> 2) * 16;
  float a = W1[(size_t)k * NH + n];
  float b = W1[(size_t)(k + 1) * NH + n];
  hi[t] = pack_hi(a, b);
  lo[t] = pack_lo(a, b);
}

// W2 padded from 10 to 16 columns with zeros
__global__ __launch_bounds__(256) void prep_w2(const float* __restrict__ W2,
                                               u32* __restrict__ hi,
                                               u32* __restrict__ lo) {
  int t = blockIdx.x * 256 + threadIdx.x;        // 16384 threads
  int kc = t >> 8, pos = t & 255;
  int lane = pos >> 3, v = pos & 7;
  int n = lane & 15;
  int k = kc * 32 + (v & 3) * 2 + (lane >> 4) * 8 + (v >> 2) * 16;
  float a = (n < NOUT) ? W2[(size_t)k * NOUT + n] : 0.0f;
  float b = (n < NOUT) ? W2[(size_t)(k + 1) * NOUT + n] : 0.0f;
  hi[t] = pack_hi(a, b);
  lo[t] = pack_lo(a, b);
}

// ---------------- fused main kernel -----------------------------------------
// grid = 1024 WGs x 128 threads (4 waves). Wave w owns rows g*128 + w*32 + 0..31.
#define JROT(P, Q, R, S)                                                       \
  do {                                                                         \
    float apq = Am[P][Q], c_, s_;                                              \
    if (__builtin_fabsf(apq) > 1e-28f) {                                       \
      float th = (Am[Q][Q] - Am[P][P]) / (2.0f * apq);                         \
      float t_ = copysignf(1.0f, th) /                                         \
                 (__builtin_fabsf(th) + __builtin_sqrtf(th * th + 1.0f));      \
      c_ = 1.0f / __builtin_sqrtf(t_ * t_ + 1.0f);                             \
      s_ = t_ * c_;                                                            \
    } else { c_ = 1.0f; s_ = 0.0f; }                                           \
    float app = Am[P][P], aqq = Am[Q][Q];                                      \
    Am[P][P] = c_ * c_ * app - 2.0f * s_ * c_ * apq + s_ * s_ * aqq;           \
    Am[Q][Q] = s_ * s_ * app + 2.0f * s_ * c_ * apq + c_ * c_ * aqq;           \
    Am[P][Q] = 0.0f; Am[Q][P] = 0.0f;                                          \
    { float x1 = Am[R][P], x2 = Am[R][Q];                                      \
      Am[R][P] = Am[P][R] = c_ * x1 - s_ * x2;                                 \
      Am[R][Q] = Am[Q][R] = s_ * x1 + c_ * x2; }                               \
    { float x1 = Am[S][P], x2 = Am[S][Q];                                      \
      Am[S][P] = Am[P][S] = c_ * x1 - s_ * x2;                                 \
      Am[S][Q] = Am[Q][S] = s_ * x1 + c_ * x2; }                               \
    { float x1 = Vm[0][P], x2 = Vm[0][Q];                                      \
      Vm[0][P] = c_ * x1 - s_ * x2; Vm[0][Q] = s_ * x1 + c_ * x2; }            \
    { float x1 = Vm[1][P], x2 = Vm[1][Q];                                      \
      Vm[1][P] = c_ * x1 - s_ * x2; Vm[1][Q] = s_ * x1 + c_ * x2; }            \
    { float x1 = Vm[2][P], x2 = Vm[2][Q];                                      \
      Vm[2][P] = c_ * x1 - s_ * x2; Vm[2][Q] = s_ * x1 + c_ * x2; }            \
    { float x1 = Vm[3][P], x2 = Vm[3][Q];                                      \
      Vm[3][P] = c_ * x1 - s_ * x2; Vm[3][Q] = s_ * x1 + c_ * x2; }            \
  } while (0)

__global__ __launch_bounds__(128, 1) void quatnet_main(
    const float* __restrict__ x, const u32* __restrict__ w1h,
    const u32* __restrict__ w1l, const u32* __restrict__ w2h,
    const u32* __restrict__ w2l, const float* __restrict__ b1,
    const float* __restrict__ b2, float* __restrict__ out) {
  extern __shared__ u32 smem[];
  const int tid = threadIdx.x;
  const int w = tid >> 5, lane = tid & 31;
  const int m0 = blockIdx.x * 128 + w * 32;

  u32*   aReg = smem + w * LDS_A_PER_WAVE;
  float* stg  = (float*)(smem + LDS_STG_BASE + w * 1024);
  float* c2s  = (float*)(smem + LDS_C2_BASE + w * 512);

  // ---- stage 1: stream x (once, NT), split to bf16 hi/lo, store pre-swizzled
  // A-fragment layout (16-bit A 16x32 per ISA: lane m holds K pairs; VGPR v ->
  // K = (v%4)*2 + kh*8 + (v/4)*16, kh = lane/16).
  for (int rr = 0; rr < 32; ++rr) {
    const int mm = rr & 15, mi = rr >> 4;
    const f32x4* xr = (const f32x4*)(x + (size_t)(m0 + rr) * ND);
#pragma unroll
    for (int pass = 0; pass < 4; ++pass) {
      f32x4 v4 = __builtin_nontemporal_load(&xr[pass * 32 + lane]);
      int kk = pass * 128 + lane * 4;          // first K of this float4
      int kc = kk >> 5, r = kk & 31;
      int v = (r >> 4) * 4 + ((r & 7) >> 1);   // (r&15)>>3 == kh below
      int lane_t = ((r & 15) >> 3) * 16 + mm;
      int idxH = (((mi * 16 + kc) * 2 + 0) * 32 + lane_t) * 8 + v;
      uint2 h; h.x = pack_hi(v4.x, v4.y); h.y = pack_hi(v4.z, v4.w);
      uint2 l; l.x = pack_lo(v4.x, v4.y); l.y = pack_lo(v4.z, v4.w);
      *(uint2*)&aReg[idxH]       = h;          // part=0 (hi)
      *(uint2*)&aReg[idxH + 256] = l;          // part=1 (lo), +32*8 dwords
    }
  }
  wave_lds_fence();

  // ---- stage 2: fused GEMM1 (split-bf16, 3 products) + relu + GEMM2 --------
  v8f c2[2] = {{}, {}};
  for (int chunk = 0; chunk < 64; ++chunk) {
    const int nt0 = chunk * 2;
    if (chunk + 1 < 64) {   // warm next chunk's first B tiles (global_prefetch_b8)
      __builtin_prefetch(w1h + ((size_t)(nt0 + 2) * 32 + lane) * 8, 0, 1);
      __builtin_prefetch(w1l + ((size_t)(nt0 + 2) * 32 + lane) * 8, 0, 1);
    }
    v8f c1[2][2] = {{{}, {}}, {{}, {}}};
    for (int kc = 0; kc < 16; ++kc) {
      u32x8 af[2][2], bf_[2][2];
#pragma unroll
      for (int mi = 0; mi < 2; ++mi)
#pragma unroll
        for (int p = 0; p < 2; ++p)
          af[mi][p] = *(const u32x8*)&aReg[(((mi * 16 + kc) * 2 + p) * 32 + lane) * 8];
#pragma unroll
      for (int ni = 0; ni < 2; ++ni) {
        size_t fo = ((size_t)(kc * 128 + nt0 + ni) * 32 + lane) * 8;
        bf_[ni][0] = *(const u32x8*)(w1h + fo);
        bf_[ni][1] = *(const u32x8*)(w1l + fo);
      }
#pragma unroll
      for (int mi = 0; mi < 2; ++mi)
#pragma unroll
        for (int ni = 0; ni < 2; ++ni) {
          c1[mi][ni] = wmma_bf16(af[mi][0], bf_[ni][0], c1[mi][ni]); // hi*hi
          c1[mi][ni] = wmma_bf16(af[mi][0], bf_[ni][1], c1[mi][ni]); // hi*lo
          c1[mi][ni] = wmma_bf16(af[mi][1], bf_[ni][0], c1[mi][ni]); // lo*hi
        }
    }
    // bias + relu, spill 32x32 tile to wave-private LDS stage (C layout:
    // vgpr r, lane -> m = r + 8*(lane/16), n = lane%16)
    float bb0 = b1[chunk * 32 + (lane & 15)];
    float bb1 = b1[chunk * 32 + 16 + (lane & 15)];
    const int nlo = lane & 15, mofs = 8 * (lane >> 4);
#pragma unroll
    for (int r = 0; r < 8; ++r) {
      stg[(r + mofs) * 32 + nlo]           = __builtin_fmaxf(c1[0][0][r] + bb0, 0.f);
      stg[(r + mofs) * 32 + 16 + nlo]      = __builtin_fmaxf(c1[0][1][r] + bb1, 0.f);
      stg[(16 + r + mofs) * 32 + nlo]      = __builtin_fmaxf(c1[1][0][r] + bb0, 0.f);
      stg[(16 + r + mofs) * 32 + 16 + nlo] = __builtin_fmaxf(c1[1][1][r] + bb1, 0.f);
    }
    wave_lds_fence();
    // GEMM2: re-read the hidden tile as split-bf16 A fragments (K = 32 hidden)
    size_t w2o = ((size_t)chunk * 32 + lane) * 8;
    u32x8 b2h = *(const u32x8*)(w2h + w2o);
    u32x8 b2l = *(const u32x8*)(w2l + w2o);
#pragma unroll
    for (int mi = 0; mi < 2; ++mi) {
      const int m = mi * 16 + (lane & 15), kh = lane >> 4;
      u32x8 a2h, a2l;
#pragma unroll
      for (int v = 0; v < 8; ++v) {
        int k2 = (v & 3) * 2 + kh * 8 + (v >> 2) * 16;
        f32x2 p = *(const f32x2*)&stg[m * 32 + k2];
        a2h[v] = pack_hi(p.x, p.y);
        a2l[v] = pack_lo(p.x, p.y);
      }
      c2[mi] = wmma_bf16(a2h, b2h, c2[mi]);
      c2[mi] = wmma_bf16(a2h, b2l, c2[mi]);
      c2[mi] = wmma_bf16(a2l, b2h, c2[mi]);
    }
    wave_lds_fence();  // protect stage reuse by next chunk's stores
  }

  // ---- stage 3: + b2, park A_vec tile, per-lane 4x4 Jacobi eigensolve ------
  const int nlo = lane & 15, mofs = 8 * (lane >> 4);
  float bv = (nlo < NOUT) ? b2[nlo] : 0.0f;
#pragma unroll
  for (int r = 0; r < 8; ++r) {
    c2s[(r + mofs) * 16 + nlo]      = c2[0][r] + bv;
    c2s[(16 + r + mofs) * 16 + nlo] = c2[1][r] + bv;
  }
  wave_lds_fence();

  float av[10];
#pragma unroll
  for (int j = 0; j < 10; ++j) av[j] = c2s[lane * 16 + j];
  float Am[4][4] = {{av[0], av[1], av[2], av[3]},
                    {av[1], av[4], av[5], av[6]},
                    {av[2], av[5], av[7], av[8]},
                    {av[3], av[6], av[8], av[9]}};
  float Vm[4][4] = {{1, 0, 0, 0}, {0, 1, 0, 0}, {0, 0, 1, 0}, {0, 0, 0, 1}};
#pragma unroll
  for (int sweep = 0; sweep < 8; ++sweep) {
    JROT(0, 1, 2, 3); JROT(0, 2, 1, 3); JROT(0, 3, 1, 2);
    JROT(1, 2, 0, 3); JROT(1, 3, 0, 2); JROT(2, 3, 0, 1);
  }
  int idx = 0; float dm = Am[0][0];
  if (Am[1][1] < dm) { dm = Am[1][1]; idx = 1; }
  if (Am[2][2] < dm) { dm = Am[2][2]; idx = 2; }
  if (Am[3][3] < dm) { dm = Am[3][3]; idx = 3; }
  float q0 = (idx == 0) ? Vm[0][0] : (idx == 1) ? Vm[0][1] : (idx == 2) ? Vm[0][2] : Vm[0][3];
  float q1 = (idx == 0) ? Vm[1][0] : (idx == 1) ? Vm[1][1] : (idx == 2) ? Vm[1][2] : Vm[1][3];
  float q2 = (idx == 0) ? Vm[2][0] : (idx == 1) ? Vm[2][1] : (idx == 2) ? Vm[2][2] : Vm[2][3];
  float q3 = (idx == 0) ? Vm[3][0] : (idx == 1) ? Vm[3][1] : (idx == 2) ? Vm[3][2] : Vm[3][3];
  float sg = (q3 < 0.0f) ? -1.0f : 1.0f;
  size_t orow = (size_t)(m0 + lane) * 4;
  out[orow + 0] = q0 * sg;
  out[orow + 1] = q1 * sg;
  out[orow + 2] = q2 * sg;
  out[orow + 3] = q3 * sg;
}

extern "C" void kernel_launch(void* const* d_in, const int* in_sizes, int n_in,
                              void* d_out, int out_size, void* d_ws, size_t ws_size,
                              hipStream_t stream) {
  (void)in_sizes; (void)n_in; (void)out_size; (void)ws_size;
  const float* x  = (const float*)d_in[0];
  const float* W1 = (const float*)d_in[1];
  const float* b1 = (const float*)d_in[2];
  const float* W2 = (const float*)d_in[3];
  const float* b2 = (const float*)d_in[4];
  u32* ws = (u32*)d_ws;                     // needs ~4.2 MB of scratch
  u32* w1h = ws + OFF_W1H; u32* w1l = ws + OFF_W1L;
  u32* w2h = ws + OFF_W2H; u32* w2l = ws + OFF_W2L;

  prep_w1<<<W1_DW / 256, 256, 0, stream>>>(W1, w1h, w1l);
  prep_w2<<<W2_DW / 256, 256, 0, stream>>>(W2, w2h, w2l);

  (void)hipFuncSetAttribute((const void*)quatnet_main,
                            hipFuncAttributeMaxDynamicSharedMemorySize, LDS_BYTES);
  quatnet_main<<<NB / 128, 128, LDS_BYTES, stream>>>(
      x, w1h, w1l, w2h, w2l, b1, b2, (float*)d_out);
}
